// Model_41549513621541
// MI455X (gfx1250) — matrix operand order repaired and checked by reading, not compiled, not run on hardware
//
#include <hip/hip_runtime.h>
#include <hip/hip_bf16.h>

typedef __bf16 bf16_t;
typedef bf16_t bf16x8  __attribute__((ext_vector_type(8)));
typedef bf16_t bf16x16 __attribute__((ext_vector_type(16)));
typedef float  f32x8   __attribute__((ext_vector_type(8)));

#define B_  64
#define T_  512
#define D_  256
#define H_  1024
#define G_  4096   // 4*H
#define O_  256
#define NGATE 16   // gate-GEMM blocks (one per WGP)
#define NPROJ 8    // projection blocks
#define NBLK  (NGATE + NPROJ)

__device__ __forceinline__ float sigf(float x) { return 1.0f / (1.0f + __expf(-x)); }

__device__ __forceinline__ f32x8 wmma_bf16(bf16x16 a, bf16x16 b, f32x8 c) {
  // (neg_a, A, neg_b, B, c_mod, C, reuse_a, reuse_b) -> v_wmma_f32_16x16x32_bf16
  return __builtin_amdgcn_wmma_f32_16x16x32_bf16(false, a, false, b, (short)0, c, false, false);
}

// A fragment (16x32 bf16, M x K). Lane L<16 -> row M=L, K=kk..kk+15;
// lanes 16-31 same rows, K=kk+16..kk+31 (matches b_frag's K order).
__device__ __forceinline__ bf16x16 a_frag(const bf16_t* __restrict__ row, int kk, int half) {
  return *(const bf16x16*)(row + kk + half * 16);   // K = kk + half*16 + [0,16)
}

// B fragment (32x16 bf16, K x N). Lane owns column N = n0+(lane&15) == one row of
// row-major W[g][k]; lanes 0-15 hold K=kk..kk+15, lanes 16-31 hold K=kk+16..+31.
__device__ __forceinline__ bf16x16 b_frag(const bf16_t* __restrict__ row, int kk, int half) {
  return *(const bf16x16*)(row + kk + half * 16);
}

// Device-wide barrier (generation ticket). NBLK co-resident blocks.
__device__ __forceinline__ void grid_sync(unsigned* __restrict__ cnt,
                                          unsigned* __restrict__ gen) {
  __threadfence();                       // publish this block's writes (agent scope)
  __syncthreads();
  if (threadIdx.x == 0) {
    // Acquire-load gen BEFORE the arrival RMW: gen cannot advance this round
    // until every block (incl. us) incremented cnt -> g is this round's gen.
    unsigned g   = __hip_atomic_load(gen, __ATOMIC_ACQUIRE, __HIP_MEMORY_SCOPE_AGENT);
    unsigned arr = __hip_atomic_fetch_add(cnt, 1u, __ATOMIC_ACQ_REL, __HIP_MEMORY_SCOPE_AGENT);
    if (arr == NBLK - 1u) {
      __hip_atomic_store(cnt, 0u, __ATOMIC_RELAXED, __HIP_MEMORY_SCOPE_AGENT);
      __hip_atomic_fetch_add(gen, 1u, __ATOMIC_RELEASE, __HIP_MEMORY_SCOPE_AGENT);
    } else {
      while (__hip_atomic_load(gen, __ATOMIC_ACQUIRE, __HIP_MEMORY_SCOPE_AGENT) == g)
        __builtin_amdgcn_s_sleep(8);     // stay off the SALU while spinning
    }
  }
  __syncthreads();
}

// One 16x16 tile of out[:, tcol, :] = h @ Wout^T + b_out
__device__ __forceinline__ void proj_one(const bf16_t* __restrict__ h,
                                         const bf16_t* __restrict__ Wout,
                                         const float* __restrict__ b_out,
                                         float* __restrict__ out, int tcol,
                                         int mtile, int ntile, int half, int ln) {
  f32x8 acc = {};
  const bf16_t* ar = h    + (size_t)(mtile * 16 + ln) * H_;
  const bf16_t* br = Wout + (size_t)(ntile * 16 + ln) * H_;
  for (int kk = 0; kk < H_; kk += 32)
    acc = wmma_bf16(a_frag(ar, kk, half), b_frag(br, kk, half), acc);
  const int   nc = ntile * 16 + ln;
  const float bo = b_out[nc];
#pragma unroll
  for (int r = 0; r < 8; ++r) {
    const int bb = mtile * 16 + half * 8 + r;              // batch row (C layout)
    out[((size_t)bb * T_ + tcol) * O_ + nc] = acc[r] + bo;
  }
}

// Persistent LSTM, heterogeneous roles:
//   blocks [0,16):  gates = bias + x_t@Wih^T + h_{t-1}@Whh^T (bf16 WMMA, f32 acc)
//                   + lane-local c/h update on the WMMA C layout
//   blocks [16,24): out[:, t-1, :] = h_{t-1}@Wout^T + b_out (hides under gate GEMM)
// One launch runs all 512 steps; steps are separated by the grid barrier.
__global__ __launch_bounds__(256)
void lstm_persist(const bf16_t* __restrict__ Xbf, const bf16_t* __restrict__ Wih,
                  const bf16_t* __restrict__ Whh, const float* __restrict__ bias,
                  const bf16_t* __restrict__ Wout, const float* __restrict__ b_out,
                  bf16_t* __restrict__ h0, bf16_t* __restrict__ h1,
                  float* __restrict__ cst, float* __restrict__ out,
                  unsigned* __restrict__ bar_cnt, unsigned* __restrict__ bar_gen)
{
  const int lane = threadIdx.x & 31;
  const int wave = threadIdx.x >> 5;   // 0..7
  const int half = lane >> 4;          // 0/1
  const int ln   = lane & 15;

  if (blockIdx.x < NGATE) {
    // ---- gate role: this wave -> batch tiles {mt0,mt0+1}, h-cols [htile*16,+16)
    const int wid   = blockIdx.x * 8 + wave;   // 0..127
    const int htile = wid >> 1;                // 0..63
    const int mt0   = (wid & 1) * 2;           // 0 or 2
    const int col   = htile * 16 + ln;         // h-column this lane owns (N index)

    const bf16_t* wih[4]; const bf16_t* whh[4];
#pragma unroll
    for (int q = 0; q < 4; ++q) {              // gate order i,f,g,o
      const int grow = q * H_ + col;
      wih[q] = Wih + (size_t)grow * D_;
      whh[q] = Whh + (size_t)grow * H_;
    }
    const float bi  = bias[col];
    const float bff = bias[H_ + col];
    const float bg  = bias[2 * H_ + col];
    const float boo = bias[3 * H_ + col];

    for (int t = 0; t < T_; ++t) {
      const bf16_t* hprev = (t & 1) ? h1 : h0;
      bf16_t*       hnext = (t & 1) ? h0 : h1;

      f32x8 acc0[4] = {}; f32x8 acc1[4] = {};

      // phase 1: x_t contribution, K = D = 256
      const bf16_t* xt  = Xbf + (size_t)t * (B_ * D_);
      const bf16_t* ax0 = xt + (size_t)(mt0 * 16 + ln) * D_;
      const bf16_t* ax1 = xt + (size_t)((mt0 + 1) * 16 + ln) * D_;
      for (int kk = 0; kk < D_; kk += 32) {
        bf16x16 a0 = a_frag(ax0, kk, half);
        bf16x16 a1 = a_frag(ax1, kk, half);
#pragma unroll
        for (int q = 0; q < 4; ++q) {
          bf16x16 b = b_frag(wih[q], kk, half);
          acc0[q] = wmma_bf16(a0, b, acc0[q]);
          acc1[q] = wmma_bf16(a1, b, acc1[q]);
        }
      }

      // phase 2: h_{t-1} contribution, K = H = 1024 (Whh streams from L2)
      const bf16_t* ah0 = hprev + (size_t)(mt0 * 16 + ln) * H_;
      const bf16_t* ah1 = hprev + (size_t)((mt0 + 1) * 16 + ln) * H_;
      for (int kk = 0; kk < H_; kk += 32) {
        bf16x16 a0 = a_frag(ah0, kk, half);
        bf16x16 a1 = a_frag(ah1, kk, half);
#pragma unroll
        for (int q = 0; q < 4; ++q) {
          bf16x16 b = b_frag(whh[q], kk, half);
          acc0[q] = wmma_bf16(a0, b, acc0[q]);
          acc1[q] = wmma_bf16(a1, b, acc1[q]);
        }
      }

      // LSTM cell update, lane-local on the C layout: VGPR r -> M=8*half+r, N=ln
#pragma unroll
      for (int mp = 0; mp < 2; ++mp) {
        const f32x8* ai = mp ? acc1 : acc0;
#pragma unroll
        for (int r = 0; r < 8; ++r) {
          const int    bb  = (mt0 + mp) * 16 + half * 8 + r;   // batch row
          const size_t idx = (size_t)bb * H_ + col;
          const float iv = sigf (ai[0][r] + bi);
          const float fv = sigf (ai[1][r] + bff);
          const float gv = tanhf(ai[2][r] + bg);
          const float ov = sigf (ai[3][r] + boo);
          const float cN = fv * cst[idx] + iv * gv;            // c owned by this lane
          cst[idx]   = cN;
          hnext[idx] = (bf16_t)(ov * tanhf(cN));
        }
      }

      grid_sync(bar_cnt, bar_gen);   // h_t visible device-wide before step t+1
    }
  } else {
    // ---- projection role: 64 waves, one 16x16 out-tile each
    const int p     = blockIdx.x - NGATE;      // 0..7
    const int widp  = p * 8 + wave;            // 0..63
    const int ntile = widp & 15;               // 0..15 (O/16)
    const int mtile = widp >> 4;               // 0..3  (B/16)

    for (int t = 0; t < T_; ++t) {
      const bf16_t* hprev = (t & 1) ? h1 : h0;
      if (t > 0)
        proj_one(hprev, Wout, b_out, out, t - 1, mtile, ntile, half, ln);
      grid_sync(bar_cnt, bar_gen);
    }
    // final projection: h_{511} lives in h0 (t=511 odd -> hnext = h0)
    proj_one(h0, Wout, b_out, out, T_ - 1, mtile, ntile, half, ln);
  }
}

// ---- prep kernels -------------------------------------------------------
__global__ void k_cvt(const float* __restrict__ in, bf16_t* __restrict__ o, int n) {
  int i = blockIdx.x * 256 + threadIdx.x;
  if (i < n) o[i] = (bf16_t)in[i];
}
// x[b][t][d] (fp32) -> Xbf[t][b][d] (bf16): per-step A rows become contiguous
__global__ void k_cvt_x(const float* __restrict__ x, bf16_t* __restrict__ xbf) {
  int i = blockIdx.x * 256 + threadIdx.x;
  if (i < B_ * T_ * D_) {
    int d  = i & (D_ - 1);
    int tt = (i >> 8) & (T_ - 1);
    int b  = i >> 17;
    xbf[((size_t)tt * B_ + b) * D_ + d] = (bf16_t)x[i];
  }
}
__global__ void k_prep(const float* __restrict__ bi, const float* __restrict__ bh,
                       float* __restrict__ bias, float* __restrict__ cst,
                       bf16_t* __restrict__ h0, unsigned* __restrict__ bar) {
  int i = blockIdx.x * 256 + threadIdx.x;
  if (i < G_) bias[i] = bi[i] + bh[i];
  if (i < B_ * H_) { cst[i] = 0.0f; h0[i] = (bf16_t)0.0f; }
  if (i < 2) bar[i] = 0u;                    // {cnt, gen}
}

extern "C" void kernel_launch(void* const* d_in, const int* in_sizes, int n_in,
                              void* d_out, int out_size, void* d_ws, size_t ws_size,
                              hipStream_t stream) {
  const float* x     = (const float*)d_in[0];
  const float* W_ih  = (const float*)d_in[1];
  const float* W_hh  = (const float*)d_in[2];
  const float* b_ih  = (const float*)d_in[3];
  const float* b_hh  = (const float*)d_in[4];
  const float* W_out = (const float*)d_in[5];
  const float* b_out = (const float*)d_in[6];
  float* out = (float*)d_out;

  // workspace carve-up (~27 MB total; all offsets 256B-aligned; L2-resident)
  char* w = (char*)d_ws;
  bf16_t*   Xbf  = (bf16_t*)w;   w += (size_t)T_ * B_ * D_ * 2;   // 16 MB
  bf16_t*   Wih  = (bf16_t*)w;   w += (size_t)G_ * D_ * 2;        //  2 MB
  bf16_t*   Whh  = (bf16_t*)w;   w += (size_t)G_ * H_ * 2;        //  8 MB
  bf16_t*   Wo   = (bf16_t*)w;   w += (size_t)O_ * H_ * 2;        // .5 MB
  float*    bias = (float*)w;    w += (size_t)G_ * 4;
  float*    cst  = (float*)w;    w += (size_t)B_ * H_ * 4;
  bf16_t*   h0   = (bf16_t*)w;   w += (size_t)B_ * H_ * 2;        // ping
  bf16_t*   h1   = (bf16_t*)w;   w += (size_t)B_ * H_ * 2;        // pong
  unsigned* bar  = (unsigned*)w; // [0]=cnt, [1]=gen

  k_cvt  <<<(G_ * D_ + 255) / 256, 256, 0, stream>>>(W_ih, Wih, G_ * D_);
  k_cvt  <<<(G_ * H_ + 255) / 256, 256, 0, stream>>>(W_hh, Whh, G_ * H_);
  k_cvt  <<<(O_ * H_ + 255) / 256, 256, 0, stream>>>(W_out, Wo, O_ * H_);
  k_cvt_x<<<(B_ * T_ * D_ + 255) / 256, 256, 0, stream>>>(x, Xbf);
  k_prep <<<(B_ * H_ + 255) / 256, 256, 0, stream>>>(b_ih, b_hh, bias, cst, h0, bar);

  // one persistent launch runs the whole recurrence + overlapped projection
  lstm_persist<<<NBLK, 256, 0, stream>>>(Xbf, Wih, Whh, bias, Wo, b_out,
                                         h0, h1, cst, out, bar, bar + 1);
}
